// IDPTransforme1r_6734508720690
// MI455X (gfx1250) — matrix operand order repaired and checked by reading, not compiled, not run on hardware
//
#include <hip/hip_runtime.h>
#include <hip/hip_bf16.h>

// ---------------------------------------------------------------------------
// Model constants (match reference)
// ---------------------------------------------------------------------------
#define DIMX   512
#define HEADSX 8
#define DHX    64
#define NBLK   6
#define DFFX   2048
#define BBX    4
#define TTX    2048
#define ROWSX  (BBX * TTX)          // 8192
#define EPSX   1e-5f

typedef _Float16 h8   __attribute__((ext_vector_type(8)));
typedef _Float16 v16h __attribute__((ext_vector_type(16)));
typedef float    v8f  __attribute__((ext_vector_type(8)));

#define WMMA_F16(a, b, c) \
    __builtin_amdgcn_wmma_f32_16x16x32_f16(false, (a), false, (b), (short)0, (c), false, false)

// Build a 16-f16 WMMA fragment from two 16-byte chunks.
// A-frag (16x32, K=32): lane half h covers K = {h*8..h*8+7, 16+h*8..16+h*8+7}
// B-frag (32x16):       lane half h covers K = {h*8..h*8+7, 16+h*8..16+h*8+7} (same as A)
__device__ __forceinline__ v16h frag16(const _Float16* p, int o0, int o1) {
    h8 x = *(const h8*)(p + o0);
    h8 y = *(const h8*)(p + o1);
    v16h r;
#pragma unroll
    for (int j = 0; j < 8; ++j) { r[j] = x[j]; r[j + 8] = y[j]; }
    return r;
}

// ---------------------------------------------------------------------------
// Weight convert + transpose: W fp32 [nb][K][N] -> Wt f16 [nb][N][K]
// ---------------------------------------------------------------------------
__global__ void wcvt_kernel(const float* __restrict__ W, _Float16* __restrict__ Wt,
                            int K, int N, int nb) {
    size_t total = (size_t)nb * K * N;
    for (size_t idx = (size_t)blockIdx.x * blockDim.x + threadIdx.x; idx < total;
         idx += (size_t)gridDim.x * blockDim.x) {
        size_t mat = idx / ((size_t)K * N);
        size_t rem = idx % ((size_t)K * N);
        int k = (int)(rem / N);
        int n = (int)(rem % N);
        Wt[mat * (size_t)K * N + (size_t)n * K + k] = (_Float16)W[idx];
    }
}

// ---------------------------------------------------------------------------
// Embedding + sinusoidal PE -> h32 (fp32) and h16 (f16)
// ---------------------------------------------------------------------------
__global__ __launch_bounds__(256) void embed_kernel(const int* __restrict__ x,
                                                    const float* __restrict__ emb,
                                                    const float* __restrict__ pe,
                                                    float* __restrict__ h32,
                                                    _Float16* __restrict__ h16) {
    int i   = blockIdx.x * blockDim.x + threadIdx.x;   // float4 units
    int row = i >> 7;                                  // DIM/4 = 128 per row
    int c4  = i & 127;
    int t   = row & (TTX - 1);
    int tok = x[row];
    float4 e = ((const float4*)(emb + (size_t)tok * DIMX))[c4];
    float4 p = ((const float4*)(pe + (size_t)t * DIMX))[c4];
    float4 r = make_float4(e.x + p.x, e.y + p.y, e.z + p.z, e.w + p.w);
    ((float4*)(h32 + (size_t)row * DIMX))[c4] = r;
    _Float16* hp = h16 + (size_t)row * DIMX + c4 * 4;
    hp[0] = (_Float16)r.x; hp[1] = (_Float16)r.y;
    hp[2] = (_Float16)r.z; hp[3] = (_Float16)r.w;
}

// ---------------------------------------------------------------------------
// WMMA GEMM: C[M,N] = A_f16[M,K] @ (Bt_f16[N,K])^T
// 256 threads = 8 waves in 4(m) x 2(n) grid; workgroup tile 128x128;
// wave tile 32x64 -> 8 WMMAs per K=32 step (2 A-frags, 4 shared B-frags).
// ---------------------------------------------------------------------------
enum { MODE_QKV = 0, MODE_F32 = 1, MODE_GELU16 = 2, MODE_BIAS32 = 3 };

template <int MODE>
__global__ __launch_bounds__(256) void gemm_wmma(
    const _Float16* __restrict__ A, const _Float16* __restrict__ Bt,
    int N, int K,
    float* __restrict__ Cf, _Float16* __restrict__ Ch,
    const float* __restrict__ bias,
    _Float16* __restrict__ q16, _Float16* __restrict__ k16, _Float16* __restrict__ v16) {
    __shared__ __align__(16) _Float16 As[128][40];
    __shared__ __align__(16) _Float16 Bs[128][40];
    const int lane = threadIdx.x & 31, wave = threadIdx.x >> 5;
    const int half = lane >> 4, l16 = lane & 15;
    const int wm = (wave & 3) * 32;       // wave m-offset within tile
    const int wn = (wave >> 2) * 64;      // wave n-offset within tile
    const int m0 = blockIdx.x * 128, n0 = blockIdx.y * 128;
    const int r  = threadIdx.x >> 1;      // staging row (2 threads/row)
    const int cp = (threadIdx.x & 1) * 16;
    v8f c[2][4] = {};

    for (int k0 = 0; k0 < K; k0 += 32) {
        __syncthreads();
        {
            const h8* sa = (const h8*)(A + (size_t)(m0 + r) * K + k0 + cp);
            *(h8*)&As[r][cp]     = sa[0];
            *(h8*)&As[r][cp + 8] = sa[1];
            const h8* sb = (const h8*)(Bt + (size_t)(n0 + r) * K + k0 + cp);
            *(h8*)&Bs[r][cp]     = sb[0];
            *(h8*)&Bs[r][cp + 8] = sb[1];
            if (k0 + 32 < K) {   // warm next tiles (global_prefetch_b8)
                __builtin_prefetch(A + (size_t)(m0 + r) * K + k0 + 32 + cp, 0, 1);
                __builtin_prefetch(Bt + (size_t)(n0 + r) * K + k0 + 32 + cp, 0, 1);
            }
        }
        __syncthreads();
        v16h a0 = frag16(&As[wm + l16][0],      half * 8, 16 + half * 8);
        v16h a1 = frag16(&As[wm + 16 + l16][0], half * 8, 16 + half * 8);
#pragma unroll
        for (int nt = 0; nt < 4; ++nt) {
            v16h b = frag16(&Bs[wn + nt * 16 + l16][0], half * 8, 16 + half * 8);
            c[0][nt] = WMMA_F16(a0, b, c[0][nt]);
            c[1][nt] = WMMA_F16(a1, b, c[1][nt]);
        }
    }

#pragma unroll
    for (int mi = 0; mi < 2; ++mi) {
#pragma unroll
        for (int nt = 0; nt < 4; ++nt) {
#pragma unroll
            for (int v = 0; v < 8; ++v) {
                float val = c[mi][nt][v];
                int gr = m0 + wm + mi * 16 + v + half * 8;
                int gc = n0 + wn + nt * 16 + l16;
                if constexpr (MODE == MODE_QKV) {
                    // reference reshape: col c -> (d = c/24, k = (c/8)%3, h = c%8)
                    int hh = gc & 7;
                    int kk = (gc >> 3) % 3;
                    int d  = gc / 24;
                    int bb = gr >> 11;          // / TTX
                    int t  = gr & (TTX - 1);
                    _Float16* dst = (kk == 0) ? q16 : (kk == 1) ? k16 : v16;
                    dst[(((size_t)bb * HEADSX + hh) * TTX + t) * DHX + d] = (_Float16)val;
                } else if constexpr (MODE == MODE_F32) {
                    Cf[(size_t)gr * N + gc] = val;
                } else if constexpr (MODE == MODE_GELU16) {
                    val += bias[gc];
                    float g = 0.5f * val * (1.0f + erff(val * 0.70710678118f));
                    Ch[(size_t)gr * N + gc] = (_Float16)g;
                } else {  // MODE_BIAS32
                    val += bias[gc];
                    Cf[(size_t)gr * N + gc] = val;
                }
            }
        }
    }
}

// ---------------------------------------------------------------------------
// Flash attention: 4 waves / WG, 64 query rows / WG, 32-key blocks, dh = 64.
// S tile is bounced through LDS (fp32) to convert C-layout -> A-layout, so
// row softmax is in-lane (+1 shfl_xor) and P lands directly in A-frag order.
// ---------------------------------------------------------------------------
__global__ __launch_bounds__(128) void attn_wmma(const _Float16* __restrict__ q16,
                                                 const _Float16* __restrict__ k16,
                                                 const _Float16* __restrict__ v16,
                                                 _Float16* __restrict__ ao16) {
    __shared__ __align__(16) float    Sls[4][16][36];   // per-wave S relayout (fp32)
    __shared__ __align__(16) _Float16 Vt[64][40];       // V^T for current key block
    const int lane = threadIdx.x & 31, wave = threadIdx.x >> 5;
    const int half = lane >> 4, l16 = lane & 15;
    const int nqb = TTX / 64;
    const int qb = blockIdx.x % nqb, bh = blockIdx.x / nqb;
    const int bb = bh / HEADSX, hh = bh % HEADSX;
    const _Float16* Qp = q16 + (size_t)bh * TTX * DHX;
    const _Float16* Kp = k16 + (size_t)bh * TTX * DHX;
    const _Float16* Vp = v16 + (size_t)bh * TTX * DHX;
    const int q0 = qb * 64 + wave * 16;

    const _Float16* qrow = Qp + (size_t)(q0 + l16) * DHX;
    v16h aq0 = frag16(qrow, half * 8, 16 + half * 8);            // dh 0..31
    v16h aq1 = frag16(qrow, 32 + half * 8, 48 + half * 8);       // dh 32..63

    float mrun = -3.0e38f, lrun = 0.0f;   // stats for row l16 (dup in both halves)
    v8f o[4] = {};

    for (int j0 = 0; j0 < TTX; j0 += 32) {
        // ---- S = Q K^T for 32 keys (two 16x16 C tiles) ----
        v8f s0 = {}, s1 = {};
        {
            const _Float16* kr0 = Kp + (size_t)(j0 + l16) * DHX;
            v16h b0 = frag16(kr0, half * 8, 16 + half * 8);
            v16h b1 = frag16(kr0, 32 + half * 8, 48 + half * 8);
            s0 = WMMA_F16(aq0, b0, s0);
            s0 = WMMA_F16(aq1, b1, s0);
            const _Float16* kr1 = Kp + (size_t)(j0 + 16 + l16) * DHX;
            v16h b2 = frag16(kr1, half * 8, 16 + half * 8);
            v16h b3 = frag16(kr1, 32 + half * 8, 48 + half * 8);
            s1 = WMMA_F16(aq0, b2, s1);
            s1 = WMMA_F16(aq1, b3, s1);
        }

        __syncthreads();   // previous block's Sls/Vt readers done
        // store scaled S (C-layout: lane=col, VGPR=row)
#pragma unroll
        for (int v = 0; v < 8; ++v) {
            Sls[wave][v + half * 8][l16]      = s0[v] * 0.125f;   // dh^-0.5
            Sls[wave][v + half * 8][16 + l16] = s1[v] * 0.125f;
        }
        // stage V^T (cooperative across all 4 waves)
        {
            int kk = threadIdx.x >> 2;          // key 0..31
            int d0 = (threadIdx.x & 3) * 16;    // dim 0,16,32,48
            const _Float16* vr = Vp + (size_t)(j0 + kk) * DHX + d0;
#pragma unroll
            for (int j = 0; j < 16; ++j) Vt[d0 + j][kk] = vr[j];
        }
        __syncthreads();

        // ---- read own row in A-frag column order; softmax in-lane ----
        float sv[16];
        {
            const float* rowp = &Sls[wave][l16][0];
#pragma unroll
            for (int j = 0; j < 8; ++j) {
                sv[j]     = rowp[half * 8 + j];
                sv[8 + j] = rowp[16 + half * 8 + j];
            }
        }
        float mx = sv[0];
#pragma unroll
        for (int j = 1; j < 16; ++j) mx = fmaxf(mx, sv[j]);
        mx = fmaxf(mx, __shfl_xor(mx, 16, 32));      // combine the two col-halves
        float nm  = fmaxf(mrun, mx);
        float fac = __expf(mrun - nm);
        mrun = nm;
        float rs = 0.0f;
        v16h pf;                                      // P already in A-frag order
#pragma unroll
        for (int j = 0; j < 16; ++j) {
            float p = __expf(sv[j] - nm);
            rs += p;
            pf[j] = (_Float16)p;
        }
        rs += __shfl_xor(rs, 16, 32);
        lrun = lrun * fac + rs;

        // broadcast fac to C-layout rows and rescale O
        float facv[8];
#pragma unroll
        for (int v = 0; v < 8; ++v) facv[v] = __shfl(fac, v + half * 8, 32);
#pragma unroll
        for (int nt = 0; nt < 4; ++nt)
#pragma unroll
            for (int v = 0; v < 8; ++v) o[nt][v] *= facv[v];

        // ---- O += P @ V ----
#pragma unroll
        for (int nt = 0; nt < 4; ++nt) {
            v16h vb = frag16(&Vt[nt * 16 + l16][0], half * 8, 16 + half * 8);
            o[nt] = WMMA_F16(pf, vb, o[nt]);
        }
    }

    // ---- normalize and write out as [b, t, h*64 + d] f16 ----
    float inv = 1.0f / lrun;
    float linv[8];
#pragma unroll
    for (int v = 0; v < 8; ++v) linv[v] = __shfl(inv, v + half * 8, 32);
#pragma unroll
    for (int v = 0; v < 8; ++v) {
        int t = q0 + v + half * 8;
        size_t ro = ((size_t)bb * TTX + t) * DIMX + hh * DHX;
#pragma unroll
        for (int nt = 0; nt < 4; ++nt)
            ao16[ro + nt * 16 + l16] = (_Float16)(o[nt][v] * linv[v]);
    }
}

// ---------------------------------------------------------------------------
// Residual add + LayerNorm (wave per row): y = LN(a + res) * g + b
// ---------------------------------------------------------------------------
__global__ __launch_bounds__(256) void ln_kernel(const float* __restrict__ a,
                                                 const float* __restrict__ res,
                                                 const float* __restrict__ g,
                                                 const float* __restrict__ bta,
                                                 float* __restrict__ y32,
                                                 _Float16* __restrict__ y16) {
    const int lane = threadIdx.x & 31, wave = threadIdx.x >> 5;
    const int row = blockIdx.x * 8 + wave;
    const float* pa = a + (size_t)row * DIMX;
    const float* pr = res + (size_t)row * DIMX;
    float x[16];
#pragma unroll
    for (int q = 0; q < 4; ++q) {
        float4 xa = ((const float4*)pa)[lane * 4 + q];
        float4 xr = ((const float4*)pr)[lane * 4 + q];
        x[q * 4 + 0] = xa.x + xr.x;  x[q * 4 + 1] = xa.y + xr.y;
        x[q * 4 + 2] = xa.z + xr.z;  x[q * 4 + 3] = xa.w + xr.w;
    }
    float s = 0.0f;
#pragma unroll
    for (int j = 0; j < 16; ++j) s += x[j];
    for (int off = 1; off < 32; off <<= 1) s += __shfl_xor(s, off, 32);
    float mu = s * (1.0f / DIMX);
    float vs = 0.0f;
#pragma unroll
    for (int j = 0; j < 16; ++j) { float d = x[j] - mu; vs += d * d; }
    for (int off = 1; off < 32; off <<= 1) vs += __shfl_xor(vs, off, 32);
    float rstd = rsqrtf(vs * (1.0f / DIMX) + EPSX);
#pragma unroll
    for (int q = 0; q < 4; ++q) {
        float4 gg = ((const float4*)g)[lane * 4 + q];
        float4 bb = ((const float4*)bta)[lane * 4 + q];
        float gv[4] = {gg.x, gg.y, gg.z, gg.w};
        float bv[4] = {bb.x, bb.y, bb.z, bb.w};
#pragma unroll
        for (int j = 0; j < 4; ++j) {
            int d = lane * 16 + q * 4 + j;
            float val = (x[q * 4 + j] - mu) * rstd * gv[j] + bv[j];
            y32[(size_t)row * DIMX + d] = val;
            y16[(size_t)row * DIMX + d] = (_Float16)val;
        }
    }
}

// ---------------------------------------------------------------------------
// Head projection: out[row] = dot(h[row], Whead) + bhead   (wave per row)
// ---------------------------------------------------------------------------
__global__ __launch_bounds__(256) void head_kernel(const float* __restrict__ h,
                                                   const float* __restrict__ w,
                                                   const float* __restrict__ bh,
                                                   float* __restrict__ out) {
    const int lane = threadIdx.x & 31, wave = threadIdx.x >> 5;
    const int row = blockIdx.x * 8 + wave;
    const float* pr = h + (size_t)row * DIMX;
    float s = 0.0f;
#pragma unroll
    for (int q = 0; q < 4; ++q) {
        float4 xv = ((const float4*)pr)[lane * 4 + q];
        float4 wv = ((const float4*)w)[lane * 4 + q];
        s += xv.x * wv.x + xv.y * wv.y + xv.z * wv.z + xv.w * wv.w;
    }
    for (int off = 1; off < 32; off <<= 1) s += __shfl_xor(s, off, 32);
    if (lane == 0) out[row] = s + bh[0];
}

// ---------------------------------------------------------------------------
// Workspace layout (bytes)
// ---------------------------------------------------------------------------
static constexpr size_t SZ_H32   = (size_t)ROWSX * DIMX * 4;   // 16 MB
static constexpr size_t SZ_H16   = (size_t)ROWSX * DIMX * 2;   //  8 MB
static constexpr size_t SZ_M16   = (size_t)ROWSX * DFFX * 2;   // 32 MB
static constexpr size_t OFF_H32   = 0;
static constexpr size_t OFF_Y32   = OFF_H32 + SZ_H32;
static constexpr size_t OFF_G32   = OFF_Y32 + SZ_H32;
static constexpr size_t OFF_H16   = OFF_G32 + SZ_H32;
static constexpr size_t OFF_Y16   = OFF_H16 + SZ_H16;
static constexpr size_t OFF_AO16  = OFF_Y16 + SZ_H16;
static constexpr size_t OFF_Q16   = OFF_AO16 + SZ_H16;
static constexpr size_t OFF_K16   = OFF_Q16 + SZ_H16;
static constexpr size_t OFF_V16   = OFF_K16 + SZ_H16;
static constexpr size_t OFF_M16   = OFF_V16 + SZ_H16;
static constexpr size_t OFF_WQKVT = OFF_M16 + SZ_M16;
static constexpr size_t OFF_W0T   = OFF_WQKVT + (size_t)NBLK * DIMX * 3 * DIMX * 2;
static constexpr size_t OFF_L1T   = OFF_W0T + (size_t)NBLK * DIMX * DIMX * 2;
static constexpr size_t OFF_L2T   = OFF_L1T + (size_t)NBLK * DIMX * DFFX * 2;

extern "C" void kernel_launch(void* const* d_in, const int* in_sizes, int n_in,
                              void* d_out, int out_size, void* d_ws, size_t ws_size,
                              hipStream_t stream) {
    (void)in_sizes; (void)n_in; (void)out_size; (void)ws_size;
    const int*   x     = (const int*)d_in[0];
    const float* embed = (const float*)d_in[1];
    const float* pe    = (const float*)d_in[2];
    const float* Wqkv  = (const float*)d_in[3];
    const float* W0    = (const float*)d_in[4];
    const float* g1    = (const float*)d_in[5];
    const float* b1    = (const float*)d_in[6];
    const float* g2    = (const float*)d_in[7];
    const float* b2    = (const float*)d_in[8];
    const float* Wl1   = (const float*)d_in[9];
    const float* bl1   = (const float*)d_in[10];
    const float* Wl2   = (const float*)d_in[11];
    const float* bl2   = (const float*)d_in[12];
    const float* Whead = (const float*)d_in[13];
    const float* bhead = (const float*)d_in[14];

    char* ws = (char*)d_ws;
    float*    h32   = (float*)(ws + OFF_H32);
    float*    y32   = (float*)(ws + OFF_Y32);
    float*    g32   = (float*)(ws + OFF_G32);
    _Float16* h16   = (_Float16*)(ws + OFF_H16);
    _Float16* y16   = (_Float16*)(ws + OFF_Y16);
    _Float16* ao16  = (_Float16*)(ws + OFF_AO16);
    _Float16* q16   = (_Float16*)(ws + OFF_Q16);
    _Float16* k16   = (_Float16*)(ws + OFF_K16);
    _Float16* v16   = (_Float16*)(ws + OFF_V16);
    _Float16* m16   = (_Float16*)(ws + OFF_M16);
    _Float16* wqkvT = (_Float16*)(ws + OFF_WQKVT);
    _Float16* w0T   = (_Float16*)(ws + OFF_W0T);
    _Float16* l1T   = (_Float16*)(ws + OFF_L1T);
    _Float16* l2T   = (_Float16*)(ws + OFF_L2T);

    // 1) weights -> f16, transposed to [N][K]
    wcvt_kernel<<<2048, 256, 0, stream>>>(Wqkv, wqkvT, DIMX, 3 * DIMX, NBLK);
    wcvt_kernel<<<2048, 256, 0, stream>>>(W0,   w0T,   DIMX, DIMX,     NBLK);
    wcvt_kernel<<<2048, 256, 0, stream>>>(Wl1,  l1T,   DIMX, DFFX,     NBLK);
    wcvt_kernel<<<2048, 256, 0, stream>>>(Wl2,  l2T,   DFFX, DIMX,     NBLK);

    // 2) embedding + positional encoding
    embed_kernel<<<(ROWSX * 128) / 256, 256, 0, stream>>>(x, embed, pe, h32, h16);

    // 3) transformer blocks
    for (int i = 0; i < NBLK; ++i) {
        const _Float16* wq = wqkvT + (size_t)i * DIMX * 3 * DIMX;
        const _Float16* w0 = w0T   + (size_t)i * DIMX * DIMX;
        const _Float16* w1 = l1T   + (size_t)i * DIMX * DFFX;
        const _Float16* w2 = l2T   + (size_t)i * DFFX * DIMX;

        gemm_wmma<MODE_QKV><<<dim3(ROWSX / 128, (3 * DIMX) / 128), 256, 0, stream>>>(
            h16, wq, 3 * DIMX, DIMX, nullptr, nullptr, nullptr, q16, k16, v16);

        attn_wmma<<<(BBX * HEADSX) * (TTX / 64), 128, 0, stream>>>(q16, k16, v16, ao16);

        gemm_wmma<MODE_F32><<<dim3(ROWSX / 128, DIMX / 128), 256, 0, stream>>>(
            ao16, w0, DIMX, DIMX, g32, nullptr, nullptr, nullptr, nullptr, nullptr);

        ln_kernel<<<ROWSX / 8, 256, 0, stream>>>(g32, h32, g1 + i * DIMX, b1 + i * DIMX,
                                                 y32, y16);

        gemm_wmma<MODE_GELU16><<<dim3(ROWSX / 128, DFFX / 128), 256, 0, stream>>>(
            y16, w1, DFFX, DIMX, nullptr, m16, bl1 + (size_t)i * DFFX,
            nullptr, nullptr, nullptr);

        gemm_wmma<MODE_BIAS32><<<dim3(ROWSX / 128, DIMX / 128), 256, 0, stream>>>(
            m16, w2, DIMX, DFFX, g32, nullptr, bl2 + (size_t)i * DIMX,
            nullptr, nullptr, nullptr);

        ln_kernel<<<ROWSX / 8, 256, 0, stream>>>(g32, y32, g2 + i * DIMX, b2 + i * DIMX,
                                                 h32, h16);
    }

    // 4) classification head
    head_kernel<<<ROWSX / 8, 256, 0, stream>>>(h32, Whead, bhead, (float*)d_out);
}
